// MultiHeadAttention_71296457114212
// MI455X (gfx1250) — matrix operand order repaired and checked by reading, not compiled, not run on hardware
//
#include <hip/hip_runtime.h>
#include <hip/hip_bf16.h>

typedef __attribute__((ext_vector_type(16))) _Float16 v16h;
typedef __attribute__((ext_vector_type(8)))  float    v8f;

#define NH     16
#define SEQ    2048
#define DEPTH  64
#define DM     1024
#define NB     2

#define LDA 40   // padded f16 row stride for LDS A tile [128][32]
#define LDB 40   // padded f16 row stride for LDS B tile [64][32] (stored [n][k])

// ---- CDNA5 async global->LDS copy (ASYNCcnt path, ISA §08) ------------------
// Copies 16 bytes per lane directly into LDS, bypassing VGPRs.
// LDS byte offset = low 32 bits of the flat shared-memory address (ISA §10.2).
__device__ __forceinline__ void async_ld_b128(void* lds_ptr, const void* gbase,
                                              int voff_bytes) {
  unsigned l = (unsigned)(size_t)lds_ptr;
  asm volatile("global_load_async_to_lds_b128 %0, %1, %2"
               :: "v"(l), "v"(voff_bytes), "s"(gbase)
               : "memory");
}
__device__ __forceinline__ void wait_asynccnt0() {
  asm volatile("s_wait_asynccnt 0x0" ::: "memory");
}

// ---- WMMA fragment loaders (layouts per CDNA5 ISA 7.12.2) -------------------

// A: 16x32 f16, M x K. lane L: M = L%16, lane-group g = L/16.
// half h -> K = (h<8 ? 0 : 16) + g*8 + h%8  (pairs are K-contiguous)
__device__ __forceinline__ v16h load_frag_a(const _Float16* lds, int tm) {
  const int lane = threadIdx.x & 31;
  const int g = lane >> 4;
  const int m = tm * 16 + (lane & 15);
  v16h a;
#pragma unroll
  for (int j = 0; j < 8; ++j) {
    const int kb = ((j < 4) ? 0 : 16) + g * 8 + (j & 3) * 2;
    a[2 * j]     = lds[m * LDA + kb];
    a[2 * j + 1] = lds[m * LDA + kb + 1];
  }
  return a;
}

// B: 32x16 f16, K x N. lane L: N = L%16, group g = L/16.
// half h -> K = (h<8 ? 0 : 16) + g*8 + h%8  (pairs are K-contiguous)
__device__ __forceinline__ v16h load_frag_b(const _Float16* lds, int tn) {
  const int lane = threadIdx.x & 31;
  const int g = lane >> 4;
  const int n = tn * 16 + (lane & 15);
  v16h b;
#pragma unroll
  for (int v = 0; v < 8; ++v) {
    const int k = ((v < 4) ? 0 : 16) + g * 8 + (v & 3) * 2;
    b[2 * v]     = lds[n * LDB + k];
    b[2 * v + 1] = lds[n * LDB + k + 1];
  }
  return b;
}

#define WMMA_F16(ACC, A, B) \
  ACC = __builtin_amdgcn_wmma_f32_16x16x32_f16(false, A, false, B, (short)0, ACC, false, false)

// ---- Kernel 1: projection  out[b,h,s,d] = (X @ W^T + bias), f16 out ---------
// Block tile 128(M) x 64(N); 8 waves, 32x32 macro-tile per wave (4 WMMA/step).
__global__ __launch_bounds__(256) void proj_kernel(
    const float* __restrict__ X,     // [NB*SEQ, DM]
    const float* __restrict__ W,     // [DM, DM]  (out,in)
    const float* __restrict__ bias,  // [DM]
    _Float16* __restrict__ out)      // [NB, NH, SEQ, DEPTH]
{
  __shared__ _Float16 sA[128 * LDA];
  __shared__ _Float16 sB[64 * LDB];
  const int m0 = blockIdx.x * 128;
  const int n0 = blockIdx.y * 64;
  const int t  = threadIdx.x;
  const int w  = t >> 5;
  const int wm = w & 3;        // 4 wave-rows of 32
  const int wn = w >> 2;       // 2 wave-cols of 32
  v8f acc00 = {}, acc01 = {}, acc10 = {}, acc11 = {};

  for (int k0 = 0; k0 < DM; k0 += 32) {
    __syncthreads();
#pragma unroll
    for (int i = 0; i < 16; ++i) {   // A: 128x32 f32 -> f16
      const int e = t + i * 256;
      const int r = e >> 5, c = e & 31;
      sA[r * LDA + c] = (_Float16)X[(size_t)(m0 + r) * DM + (k0 + c)];
    }
#pragma unroll
    for (int i = 0; i < 8; ++i) {    // B: 64x32 f32 -> f16
      const int e = t + i * 256;
      const int r = e >> 5, c = e & 31;
      sB[r * LDB + c] = (_Float16)W[(size_t)(n0 + r) * DM + (k0 + c)];
    }
    __syncthreads();
    const v16h a0 = load_frag_a(sA, wm * 2);
    const v16h a1 = load_frag_a(sA, wm * 2 + 1);
    const v16h b0 = load_frag_b(sB, wn * 2);
    const v16h b1 = load_frag_b(sB, wn * 2 + 1);
    WMMA_F16(acc00, a0, b0); WMMA_F16(acc01, a0, b1);
    WMMA_F16(acc10, a1, b0); WMMA_F16(acc11, a1, b1);
  }

  const int lane = t & 31;
  const int g  = lane >> 4;
  const int nl = lane & 15;
  const v8f* accs[4] = { &acc00, &acc01, &acc10, &acc11 };
#pragma unroll
  for (int ii = 0; ii < 2; ++ii) {
#pragma unroll
    for (int jj = 0; jj < 2; ++jj) {
      const v8f& acc = *accs[ii * 2 + jj];
      const int nglob = n0 + (wn * 2 + jj) * 16 + nl;
      const float bv = bias[nglob];
      const int h = nglob >> 6, d = nglob & 63;
#pragma unroll
      for (int r2 = 0; r2 < 8; ++r2) {
        const int m = m0 + (wm * 2 + ii) * 16 + g * 8 + r2;
        const int bidx = m >> 11, s = m & (SEQ - 1);
        out[(((size_t)(bidx * NH + h)) * SEQ + s) * DEPTH + d] =
            (_Float16)(acc[r2] + bv);
      }
    }
  }
}

// ---- Kernel 2: scores = mask(Q K^T / 8), f32 into attn ----------------------
// A/B tiles are raw f16 copies -> use GLOBAL_LOAD_ASYNC_TO_LDS_B128.
__global__ __launch_bounds__(256) void scores_kernel(
    const _Float16* __restrict__ Qh,   // [NB*NH, SEQ, DEPTH]
    const _Float16* __restrict__ Kh,   // [NB*NH, SEQ, DEPTH]
    const int* __restrict__ mask,      // [SEQ, SEQ]
    float* __restrict__ attn)          // [NB*NH, SEQ, SEQ]
{
  __shared__ _Float16 sA[128 * LDA];
  __shared__ _Float16 sB[64 * LDB];
  const int m0 = blockIdx.x * 128;  // query rows
  const int n0 = blockIdx.y * 64;   // key cols
  const int bh = blockIdx.z;
  const _Float16* Q = Qh + (size_t)bh * SEQ * DEPTH + (size_t)m0 * DEPTH;
  const _Float16* K = Kh + (size_t)bh * SEQ * DEPTH + (size_t)n0 * DEPTH;
  const int t  = threadIdx.x;
  const int w  = t >> 5;
  const int wm = w & 3;
  const int wn = w >> 2;
  v8f acc00 = {}, acc01 = {}, acc10 = {}, acc11 = {};

  for (int k0 = 0; k0 < DEPTH; k0 += 32) {
    __syncthreads();
    // A: 128 rows x 32 halfs = 512 x 16B chunks; 2 per thread.
#pragma unroll
    for (int i = 0; i < 2; ++i) {
      const int c = t + i * 256;
      const int r = c >> 2, kc = c & 3;
      async_ld_b128(&sA[r * LDA + kc * 8], Q,
                    (int)((r * DEPTH + k0 + kc * 8) * 2));
    }
    // B: 64 rows x 32 halfs = 256 x 16B chunks; 1 per thread.
    {
      const int r = t >> 2, kc = t & 3;
      async_ld_b128(&sB[r * LDB + kc * 8], K,
                    (int)((r * DEPTH + k0 + kc * 8) * 2));
    }
    wait_asynccnt0();
    __syncthreads();
    const v16h a0 = load_frag_a(sA, wm * 2);
    const v16h a1 = load_frag_a(sA, wm * 2 + 1);
    const v16h b0 = load_frag_b(sB, wn * 2);
    const v16h b1 = load_frag_b(sB, wn * 2 + 1);
    WMMA_F16(acc00, a0, b0); WMMA_F16(acc01, a0, b1);
    WMMA_F16(acc10, a1, b0); WMMA_F16(acc11, a1, b1);
  }

  const int lane = t & 31;
  const int g  = lane >> 4;
  const int nl = lane & 15;
  const v8f* accs[4] = { &acc00, &acc01, &acc10, &acc11 };
#pragma unroll
  for (int ii = 0; ii < 2; ++ii) {
#pragma unroll
    for (int jj = 0; jj < 2; ++jj) {
      const v8f& acc = *accs[ii * 2 + jj];
      const int nk = n0 + (wn * 2 + jj) * 16 + nl;
#pragma unroll
      for (int r2 = 0; r2 < 8; ++r2) {
        const int mq = m0 + (wm * 2 + ii) * 16 + g * 8 + r2;
        float val = acc[r2] * 0.125f;               // 1/sqrt(64)
        if (mask[(size_t)mq * SEQ + nk] == 0) val = -1e9f;
        attn[((size_t)bh * SEQ + mq) * SEQ + nk] = val;
      }
    }
  }
}

// ---- Kernel 3: row softmax over 2048, in place ------------------------------
__global__ __launch_bounds__(256) void softmax_kernel(float* __restrict__ attn)
{
  __shared__ float red[256];
  float* p = attn + (size_t)blockIdx.x * SEQ;
  const int t = threadIdx.x;
  float v[8];
  float mx = -3.4e38f;
#pragma unroll
  for (int i = 0; i < 8; ++i) { v[i] = p[t + i * 256]; mx = fmaxf(mx, v[i]); }
  red[t] = mx; __syncthreads();
  for (int s = 128; s > 0; s >>= 1) {
    if (t < s) red[t] = fmaxf(red[t], red[t + s]);
    __syncthreads();
  }
  mx = red[0]; __syncthreads();
  float sum = 0.f;
#pragma unroll
  for (int i = 0; i < 8; ++i) { v[i] = __expf(v[i] - mx); sum += v[i]; }
  red[t] = sum; __syncthreads();
  for (int s = 128; s > 0; s >>= 1) {
    if (t < s) red[t] += red[t + s];
    __syncthreads();
  }
  const float inv = 1.0f / red[0];
#pragma unroll
  for (int i = 0; i < 8; ++i) p[t + i * 256] = v[i] * inv;
}

// ---- Kernel 4: O = P @ V, concat-heads f16 output ---------------------------
__global__ __launch_bounds__(256) void av_kernel(
    const float* __restrict__ attn,    // [NB*NH, SEQ, SEQ] probs
    const _Float16* __restrict__ Vh,   // [NB*NH, SEQ, DEPTH]
    _Float16* __restrict__ Ocat)       // [NB*SEQ, DM]
{
  __shared__ _Float16 sA[128 * LDA];
  __shared__ _Float16 sB[64 * LDB];
  const int m0 = blockIdx.x * 128;  // query rows; N = full DEPTH=64
  const int bh = blockIdx.z;
  const float* P = attn + (size_t)bh * SEQ * SEQ;
  const _Float16* V = Vh + (size_t)bh * SEQ * DEPTH;
  const int t  = threadIdx.x;
  const int w  = t >> 5;
  const int wm = w & 3;
  const int wn = w >> 2;
  v8f acc00 = {}, acc01 = {}, acc10 = {}, acc11 = {};

  for (int k0 = 0; k0 < SEQ; k0 += 32) {
    __syncthreads();
#pragma unroll
    for (int i = 0; i < 16; ++i) {   // A: 128x32 f32 probs -> f16
      const int e = t + i * 256;
      const int r = e >> 5, c = e & 31;
      sA[r * LDA + c] = (_Float16)P[(size_t)(m0 + r) * SEQ + (k0 + c)];
    }
#pragma unroll
    for (int i = 0; i < 8; ++i) {    // B: transpose V rows into [n][k]
      const int e = t + i * 256;
      const int kb = e >> 6, nb = e & 63;
      sB[nb * LDB + kb] = V[(size_t)(k0 + kb) * DEPTH + nb];
    }
    __syncthreads();
    const v16h a0 = load_frag_a(sA, wm * 2);
    const v16h a1 = load_frag_a(sA, wm * 2 + 1);
    const v16h b0 = load_frag_b(sB, wn * 2);
    const v16h b1 = load_frag_b(sB, wn * 2 + 1);
    WMMA_F16(acc00, a0, b0); WMMA_F16(acc01, a0, b1);
    WMMA_F16(acc10, a1, b0); WMMA_F16(acc11, a1, b1);
  }

  const int lane = t & 31;
  const int g  = lane >> 4;
  const int nl = lane & 15;
  const int bidx = bh >> 4, h = bh & 15;
  const v8f* accs[4] = { &acc00, &acc01, &acc10, &acc11 };
#pragma unroll
  for (int ii = 0; ii < 2; ++ii) {
#pragma unroll
    for (int jj = 0; jj < 2; ++jj) {
      const v8f& acc = *accs[ii * 2 + jj];
      const int d = (wn * 2 + jj) * 16 + nl;
#pragma unroll
      for (int r2 = 0; r2 < 8; ++r2) {
        const int s = m0 + (wm * 2 + ii) * 16 + g * 8 + r2;
        Ocat[((size_t)(bidx * SEQ + s)) * DM + h * DEPTH + d] = (_Float16)acc[r2];
      }
    }
  }
}

// ---- Kernel 5: final dense  out = Ocat @ Wd^T + bd, f32 out -----------------
// A tile is a raw f16 copy -> async-to-LDS; B tile needs f32->f16 conversion.
__global__ __launch_bounds__(256) void dense_kernel(
    const _Float16* __restrict__ Xh,  // [NB*SEQ, DM] f16
    const float* __restrict__ W,      // [DM, DM]
    const float* __restrict__ bias,   // [DM]
    float* __restrict__ out)          // [NB*SEQ, DM] f32
{
  __shared__ _Float16 sA[128 * LDA];
  __shared__ _Float16 sB[64 * LDB];
  const int m0 = blockIdx.x * 128;
  const int n0 = blockIdx.y * 64;
  const _Float16* XA = Xh + (size_t)m0 * DM;
  const int t  = threadIdx.x;
  const int w  = t >> 5;
  const int wm = w & 3;
  const int wn = w >> 2;
  v8f acc00 = {}, acc01 = {}, acc10 = {}, acc11 = {};

  for (int k0 = 0; k0 < DM; k0 += 32) {
    __syncthreads();
    // A: 128x32 halfs = 512 x 16B chunks; 2 per thread, async to LDS.
#pragma unroll
    for (int i = 0; i < 2; ++i) {
      const int c = t + i * 256;
      const int r = c >> 2, kc = c & 3;
      async_ld_b128(&sA[r * LDA + kc * 8], XA,
                    (int)((r * DM + k0 + kc * 8) * 2));
    }
#pragma unroll
    for (int i = 0; i < 8; ++i) {    // B: 64x32 f32 -> f16
      const int e = t + i * 256;
      const int r = e >> 5, c = e & 31;
      sB[r * LDB + c] = (_Float16)W[(size_t)(n0 + r) * DM + (k0 + c)];
    }
    wait_asynccnt0();
    __syncthreads();
    const v16h a0 = load_frag_a(sA, wm * 2);
    const v16h a1 = load_frag_a(sA, wm * 2 + 1);
    const v16h b0 = load_frag_b(sB, wn * 2);
    const v16h b1 = load_frag_b(sB, wn * 2 + 1);
    WMMA_F16(acc00, a0, b0); WMMA_F16(acc01, a0, b1);
    WMMA_F16(acc10, a1, b0); WMMA_F16(acc11, a1, b1);
  }

  const int lane = t & 31;
  const int g  = lane >> 4;
  const int nl = lane & 15;
  const v8f* accs[4] = { &acc00, &acc01, &acc10, &acc11 };
#pragma unroll
  for (int ii = 0; ii < 2; ++ii) {
#pragma unroll
    for (int jj = 0; jj < 2; ++jj) {
      const v8f& acc = *accs[ii * 2 + jj];
      const int nglob = n0 + (wn * 2 + jj) * 16 + nl;
      const float bv = bias[nglob];
#pragma unroll
      for (int r2 = 0; r2 < 8; ++r2) {
        const int m = m0 + (wm * 2 + ii) * 16 + g * 8 + r2;
        out[(size_t)m * DM + nglob] = acc[r2] + bv;
      }
    }
  }
}

// ---- Host-side launch -------------------------------------------------------
extern "C" void kernel_launch(void* const* d_in, const int* in_sizes, int n_in,
                              void* d_out, int out_size, void* d_ws, size_t ws_size,
                              hipStream_t stream) {
  (void)in_sizes; (void)n_in; (void)out_size; (void)ws_size;
  const float* query = (const float*)d_in[0];
  const float* key   = (const float*)d_in[1];
  const float* value = (const float*)d_in[2];
  const int*   mask  = (const int*)  d_in[3];
  const float* wq_w  = (const float*)d_in[4];
  const float* wq_b  = (const float*)d_in[5];
  const float* wk_w  = (const float*)d_in[6];
  const float* wk_b  = (const float*)d_in[7];
  const float* wv_w  = (const float*)d_in[8];
  const float* wv_b  = (const float*)d_in[9];
  const float* wd_w  = (const float*)d_in[10];
  const float* wd_b  = (const float*)d_in[11];

  float* out  = (float*)d_out;                              // [NB,SEQ,DM]
  float* attn = out + (size_t)NB * SEQ * DM;                // [NB,NH,SEQ,SEQ]

  const size_t headElems = (size_t)NB * NH * SEQ * DEPTH;   // 4M f16 each
  _Float16* Qh = (_Float16*)d_ws;
  _Float16* Kh = Qh + headElems;
  _Float16* Vh = Kh + headElems;
  _Float16* Oc = Vh + headElems;                            // [NB*SEQ, DM] f16

  const dim3 blk(256);
  proj_kernel<<<dim3(32, 16), blk, 0, stream>>>(query, wq_w, wq_b, Qh);
  proj_kernel<<<dim3(32, 16), blk, 0, stream>>>(key,   wk_w, wk_b, Kh);
  proj_kernel<<<dim3(32, 16), blk, 0, stream>>>(value, wv_w, wv_b, Vh);
  scores_kernel<<<dim3(16, 32, 32), blk, 0, stream>>>(Qh, Kh, mask, attn);
  softmax_kernel<<<dim3(NB * NH * SEQ), blk, 0, stream>>>(attn);
  av_kernel<<<dim3(16, 1, 32), blk, 0, stream>>>(attn, Vh, Oc);
  dense_kernel<<<dim3(32, 16), blk, 0, stream>>>(Oc, wd_w, wd_b, out);
}
